// Model_78151224918694
// MI455X (gfx1250) — compile-verified, matrix operand order checked by reading, not run on hardware
//
#include <hip/hip_runtime.h>
#include <hip/hip_bf16.h>
#include <math.h>

typedef __attribute__((ext_vector_type(16))) _Float16 v16h;
typedef __attribute__((ext_vector_type(8)))  _Float16 v8h;
typedef __attribute__((ext_vector_type(8)))  float    v8f;

#define BB   8
#define CC   8
#define HHn  256
#define WWn  256
#define HWs  (HHn * WWn)
#define CK   72          // C * 3 * 3 contraction depth (padded to 96)
#define SEG  128         // pixels per workgroup (8 waves x 16)
#define LDW  132         // raw tile row stride (130 used: SEG + 2 halo)
#define TAUc 10.0f
#define ALPHAc 2.0f

// ---------------------------------------------------------------------------
// Stage raw input halo tile (8 ch x 3 rows x 130 px) into LDS as f16.
// Spatial zero padding resolved here, once. absmode: value = |d1[c] - x1|.
// ---------------------------------------------------------------------------
__device__ __forceinline__ void stage_raw(const float* __restrict__ src,
                                          const float* __restrict__ sub, int absmode,
                                          int b, int h, int wseg,
                                          _Float16* __restrict__ rawT)
{
    for (int i = threadIdx.x; i < CC * 3 * (SEG + 2); i += 256) {
        int x  = i % (SEG + 2);
        int rc = i / (SEG + 2);              // c*3 + ki
        int ki = rc % 3;
        int c  = rc / 3;
        int hn = h + ki - 1, wn = wseg - 1 + x;
        float v = 0.f;
        if ((unsigned)hn < (unsigned)HHn && (unsigned)wn < (unsigned)WWn) {
            int pix = hn * WWn + wn;
            v = src[((size_t)(b * CC + c)) * HWs + pix];
            if (absmode) v = fabsf(v - sub[(size_t)b * HWs + pix]);
        }
        rawT[rc * LDW + x] = (_Float16)v;
    }
}

// B panel staged [n][128] f16, zero-padded for n>=8 and K>=72.
__device__ __forceinline__ void stage_bmat(const float* __restrict__ wgt,
                                           _Float16* __restrict__ Bld)
{
    for (int i = threadIdx.x; i < 16 * 128; i += 256) {
        int n = i >> 7, K = i & 127;
        float v = (n < CC && K < CK) ? wgt[n * CK + K] : 0.f;
        Bld[i] = (_Float16)v;
    }
}

// Load a 16x32 f16 WMMA fragment from a row-major LDS panel ([row][96] A, [row][128] B).
// Slots 0..7 -> K = 32s + 8*hi + 0..7 ; slots 8..15 -> +16. Two ds_load_b128.
__device__ __forceinline__ v16h load_frag(const _Float16* __restrict__ panel,
                                          int rowbase_halfs, int s, int hi)
{
    const _Float16* p = panel + rowbase_halfs + 32 * s + hi * 8;
    v8h a0 = *(const v8h*)(p);
    v8h a1 = *(const v8h*)(p + 16);
    v16h af;
#pragma unroll
    for (int j = 0; j < 8; ++j) { af[j] = a0[j]; af[j + 8] = a1[j]; }
    return af;
}

// ---------------------------------------------------------------------------
// 3x3 conv, 8 -> 8 ch, SAME padding: LDS im2col + WMMA implicit GEMM.
// One block per (b, h, 128-px segment); wave wv owns 16-px tile wv.
// ---------------------------------------------------------------------------
__global__ void __launch_bounds__(256)
conv3x3_wmma(const float* __restrict__ src, const float* __restrict__ wgt,
             float* __restrict__ dst, const float* __restrict__ sub, int absmode)
{
    __shared__ __align__(16) _Float16 rawT[CC * 3 * LDW];   // 6336 B
    __shared__ __align__(16) _Float16 Ald[SEG * 96];        // 24576 B
    __shared__ __align__(16) _Float16 Bld[16 * 128];        // 4096 B

    const int seg = blockIdx.x & 1;
    const int h   = (blockIdx.x >> 1) & 255;
    const int b   = blockIdx.x >> 9;
    const int wseg = seg * SEG;

    stage_raw(src, sub, absmode, b, h, wseg, rawT);
    stage_bmat(wgt, Bld);
    __syncthreads();

    // im2col: thread owns one pixel + an unrolled compile-time K range,
    // so (c,ki,kj) fold to immediates; K>=72 pad rows written as zeros.
    {
        const int px = threadIdx.x & 127;
        if (threadIdx.x < 128) {
#pragma unroll
            for (int K = 0; K < 48; ++K) {
                int c = K / 9, r = K - 9 * c, ki = r / 3, kj = r - 3 * ki;
                Ald[px * 96 + K] = rawT[(c * 3 + ki) * LDW + px + kj];
            }
        } else {
#pragma unroll
            for (int K = 48; K < 96; ++K) {
                _Float16 v = (_Float16)0.f;
                if (K < CK) {
                    int c = K / 9, r = K - 9 * c, ki = r / 3, kj = r - 3 * ki;
                    v = rawT[(c * 3 + ki) * LDW + px + kj];
                }
                Ald[px * 96 + K] = v;
            }
        }
    }
    __syncthreads();

    const int lane = threadIdx.x & 31;
    const int wv   = threadIdx.x >> 5;
    const int m    = lane & 15;
    const int hi   = lane >> 4;
    const int abase = (wv * 16 + m) * 96;

    v8f acc = {};
#pragma unroll
    for (int s = 0; s < 3; ++s) {
        v16h af = load_frag(Ald, abase, s, hi);
        v16h bf = load_frag(Bld, m * 128, s, hi);
        acc = __builtin_amdgcn_wmma_f32_16x16x32_f16(false, af, false, bf,
                                                     (short)0, acc, false, false);
    }

    // D layout: VGPR r -> M = r + 8*hi ; lane -> N = m (output channel)
    if (m < CC) {
#pragma unroll
        for (int r = 0; r < 8; ++r) {
            int M = r + hi * 8;
            dst[((size_t)(b * CC + m)) * HWs + h * WWn + wseg + wv * 16 + M] = acc[r];
        }
    }
}

// ---------------------------------------------------------------------------
// Fused: w1 = paconv(f, Wp) * f ; x1 = softargmax(d1, w1, TAU).
// Gaussian kernel computed from the LDS raw tile and folded into the im2col
// A panel, so the einsum is a plain WMMA GEMM.
// ---------------------------------------------------------------------------
__global__ void __launch_bounds__(256)
paconv_softargmax_wmma(const float* __restrict__ f, const float* __restrict__ wgt,
                       const float* __restrict__ d1, float* __restrict__ x1,
                       float* __restrict__ out, int stage)
{
    __shared__ __align__(16) _Float16 rawT[CC * 3 * LDW];
    __shared__ __align__(16) _Float16 Ald[SEG * 96];
    __shared__ __align__(16) _Float16 Bld[16 * 128];
    __shared__ float kernL[SEG * 9];
    __shared__ float w1s[SEG][8];

    const int seg = blockIdx.x & 1;
    const int h   = (blockIdx.x >> 1) & 255;
    const int b   = blockIdx.x >> 9;
    const int wseg = seg * SEG;

    stage_raw(f, nullptr, 0, b, h, wseg, rawT);
    stage_bmat(wgt, Bld);
    __syncthreads();

    // kern[px][k] = exp(-0.5 * sum_c (p - center)^2), from LDS raw tile.
    // Zero-padded neighbors use p = 0, matching the reference semantics.
    for (int i = threadIdx.x; i < SEG * 9; i += 256) {
        int px = i / 9, k = i - 9 * px;
        int ki = k / 3, kj = k - 3 * ki;
        float ss = 0.f;
#pragma unroll
        for (int c = 0; c < CC; ++c) {
            float ctr = (float)rawT[(c * 3 + 1) * LDW + px + 1];
            float nb  = (float)rawT[(c * 3 + ki) * LDW + px + kj];
            float d = nb - ctr;
            ss += d * d;
        }
        kernL[i] = expf(-0.5f * ss);
    }
    __syncthreads();

    // im2col with kern folded in: A[px][K] = p[c][k][px] * kern[px][k]
    {
        const int px = threadIdx.x & 127;
        if (threadIdx.x < 128) {
#pragma unroll
            for (int K = 0; K < 48; ++K) {
                int c = K / 9, r = K - 9 * c, ki = r / 3, kj = r - 3 * ki;
                float pv = (float)rawT[(c * 3 + ki) * LDW + px + kj];
                Ald[px * 96 + K] = (_Float16)(pv * kernL[px * 9 + r]);
            }
        } else {
#pragma unroll
            for (int K = 48; K < 96; ++K) {
                _Float16 v = (_Float16)0.f;
                if (K < CK) {
                    int c = K / 9, r = K - 9 * c, ki = r / 3, kj = r - 3 * ki;
                    float pv = (float)rawT[(c * 3 + ki) * LDW + px + kj];
                    v = (_Float16)(pv * kernL[px * 9 + r]);
                }
                Ald[px * 96 + K] = v;
            }
        }
    }
    __syncthreads();

    const int lane = threadIdx.x & 31;
    const int wv   = threadIdx.x >> 5;
    const int m    = lane & 15;
    const int hi   = lane >> 4;
    const int abase = (wv * 16 + m) * 96;

    v8f acc = {};
#pragma unroll
    for (int s = 0; s < 3; ++s) {
        v16h af = load_frag(Ald, abase, s, hi);
        v16h bf = load_frag(Bld, m * 128, s, hi);
        acc = __builtin_amdgcn_wmma_f32_16x16x32_f16(false, af, false, bf,
                                                     (short)0, acc, false, false);
    }

    // w1 = paconv * f(center); transpose through LDS for channel softmax
    if (m < CC) {
#pragma unroll
        for (int r = 0; r < 8; ++r) {
            int M = r + hi * 8;
            int pxl = wv * 16 + M;
            float fv = (float)rawT[(m * 3 + 1) * LDW + pxl + 1];
            w1s[pxl][m] = acc[r] * fv;
        }
    }
    __syncthreads();

    if (threadIdx.x < SEG) {
        int pxl = threadIdx.x;
        int pix = h * WWn + wseg + pxl;
        float t8[8];
        float mx = -1e30f;
#pragma unroll
        for (int o = 0; o < 8; ++o) {
            t8[o] = TAUc * w1s[pxl][o];
            mx = fmaxf(mx, t8[o]);
        }
        float se = 0.f, num = 0.f;
#pragma unroll
        for (int o = 0; o < 8; ++o) {
            float e = expf(t8[o] - mx);
            se  += e;
            num += e * d1[((size_t)(b * CC + o)) * HWs + pix];
        }
        float x = num / se;
        x1[(size_t)b * HWs + pix] = x;
        out[((size_t)(b * 4 + stage)) * HWs + pix] = x;
    }
}

// ---------------------------------------------------------------------------
// Grouped pixel-adaptive 2-way blend + d1 update (VALU; K=18 too small
// for WMMA). wg layout: [g=8][co=2][cg=2][k=9]. Pointwise in d1.
// ---------------------------------------------------------------------------
__global__ void __launch_bounds__(256)
update_d1_kernel(const float* __restrict__ f, const float* __restrict__ fx,
                 const float* __restrict__ d1_in, const float* __restrict__ x1,
                 const float* __restrict__ wg, float* __restrict__ d1_out)
{
    int idx = blockIdx.x * blockDim.x + threadIdx.x;     // (b, s, pix)
    int pix = idx & (HWs - 1);
    int s = (idx >> 16) & 7;
    int b = idx >> 19;
    if (b >= BB) return;
    int h = pix >> 8, w = pix & 255;

    size_t base = ((size_t)(b * CC + s)) * HWs;
    float cf  = f[base + pix];
    float cfx = fx[base + pix];
    float a0 = 0.f, a1 = 0.f;
#pragma unroll
    for (int k = 0; k < 9; ++k) {
        int ki = k / 3, kj = k % 3;
        int hn = h + ki - 1, wn = w + kj - 1;
        float pf = 0.f, pfx = 0.f;
        if ((unsigned)hn < (unsigned)HHn && (unsigned)wn < (unsigned)WWn) {
            pf  = f[base + hn * WWn + wn];
            pfx = fx[base + hn * WWn + wn];
        }
        float df = pf - cf, dfx = pfx - cfx;
        float kern = expf(-0.5f * (df * df + dfx * dfx));
        a0 += kern * (wg[((s * 2 + 0) * 2 + 0) * 9 + k] * pf +
                      wg[((s * 2 + 0) * 2 + 1) * 9 + k] * pfx);
        a1 += kern * (wg[((s * 2 + 1) * 2 + 0) * 9 + k] * pf +
                      wg[((s * 2 + 1) * 2 + 1) * 9 + k] * pfx);
    }
    a0 *= ALPHAc; a1 *= ALPHAc;
    float mx = fmaxf(a0, a1);
    float e0 = expf(a0 - mx), e1 = expf(a1 - mx);
    float dv = d1_in[base + pix];
    float xv = x1[(size_t)b * HWs + pix];
    d1_out[base + pix] = (e0 * dv + e1 * xv) / (e0 + e1);
}

// ---------------------------------------------------------------------------
extern "C" void kernel_launch(void* const* d_in, const int* in_sizes, int n_in,
                              void* d_out, int out_size, void* d_ws, size_t ws_size,
                              hipStream_t stream)
{
    (void)in_sizes; (void)n_in; (void)out_size; (void)ws_size;

    const float* d1in = (const float*)d_in[0];   // (8,8,256,256)
    const float* We   = (const float*)d_in[1];   // (4,8,8,3,3)  stride 576
    const float* Wp   = (const float*)d_in[2];   // (4,8,8,3,3)
    const float* We2  = (const float*)d_in[3];   // (3,8,8,3,3)
    const float* Wg   = (const float*)d_in[4];   // (3,16,2,3,3) stride 288
    float* out = (float*)d_out;                  // (8,4,256,256)

    const size_t NCH = (size_t)BB * CC * HWs;    // 4.19M floats
    float* ws     = (float*)d_ws;
    float* f_buf  = ws;
    float* fx_buf = ws + NCH;
    float* d1_buf = ws + 2 * NCH;
    float* x1_buf = ws + 3 * NCH;

    const int convBlocks = BB * HHn * (WWn / SEG);            // 4096
    const int updBlocks  = (int)((size_t)BB * CC * HWs / 256); // 16384

    const float* d1cur = d1in;
    for (int i = 0; i < 4; ++i) {
        conv3x3_wmma<<<convBlocks, 256, 0, stream>>>(
            d1cur, We + i * 576, f_buf, nullptr, 0);
        paconv_softargmax_wmma<<<convBlocks, 256, 0, stream>>>(
            f_buf, Wp + i * 576, d1cur, x1_buf, out, i);
        if (i < 3) {
            conv3x3_wmma<<<convBlocks, 256, 0, stream>>>(
                d1cur, We2 + i * 576, fx_buf, x1_buf, 1);
            update_d1_kernel<<<updBlocks, 256, 0, stream>>>(
                f_buf, fx_buf, d1cur, x1_buf, Wg + i * 288, d1_buf);
            d1cur = d1_buf;
        }
    }
}
